// SparseRegisterStep_3865470566779
// MI455X (gfx1250) — matrix operand order repaired and checked by reading, not compiled, not run on hardware
//
#include <hip/hip_runtime.h>
#include <hip/hip_bf16.h>
#include <math.h>

typedef __attribute__((ext_vector_type(16))) _Float16 v16h;
typedef __attribute__((ext_vector_type(8)))  float    v8f;
typedef __attribute__((ext_vector_type(4)))  unsigned int tdm_u32x4;
typedef __attribute__((ext_vector_type(8)))  int          tdm_i32x8;
typedef __attribute__((ext_vector_type(4)))  int          tdm_i32x4;
typedef _Float16 h16;

#define EPS_RMS 1.1920929e-07f

// ---------------------------------------------------------------------------
// WMMA wrapper: D = A(16x32 f16) * B(32x16 f16) + C(16x16 f32)
// ---------------------------------------------------------------------------
__device__ __forceinline__ v8f wmma_f16(v16h a, v16h b, v8f c) {
  return __builtin_amdgcn_wmma_f32_16x16x32_f16(
      /*neg_a=*/false, a, /*neg_b=*/false, b,
      /*c_mod=*/(short)0, c, /*reuse_a=*/false, /*reuse_b=*/false);
}

// A-fragment (16x32, 16-bit): lane m=L&15 holds row M=m; half h=L>>4 selects K
// groups. VGPR v holds K pair at kb = h*16 + v*2.
__device__ __forceinline__ v16h load_a_tile(const h16* __restrict__ A, int ld,
                                            int row0, int k0, int lane) {
  int m = lane & 15, h = lane >> 4;
  const h16* r = A + (size_t)(row0 + m) * ld + k0;
  v16h a;
#pragma unroll
  for (int v = 0; v < 8; ++v) {
    int kb = h * 16 + v * 2;
    a[2 * v]     = r[kb];
    a[2 * v + 1] = r[kb + 1];
  }
  return a;
}

// B-fragment (32x16, 16-bit) built from a ROW-MAJOR matrix W[N][Kd] used as
// W^T: lane n=L&15 is column N=col0+n, half h=L>>4; element e holds
// K = h*16+e, i.e. 16 contiguous halves from W's row.
__device__ __forceinline__ v16h load_bt_tile(const h16* __restrict__ W, int ld,
                                             int col0, int k0, int lane) {
  int n = lane & 15, h = lane >> 4;
  const h16* r = W + (size_t)(col0 + n) * ld + k0 + h * 16;
  v16h b;
#pragma unroll
  for (int e = 0; e < 16; ++e) b[e] = r[e];
  return b;
}

// ---------------------------------------------------------------------------
// Tensor Data Mover: DMA `rows` x `cols` (f16, row stride `ld` elements) from
// global memory into LDS at byte offset lds_addr. Tracked with TENSORcnt.
// D# packing per CDNA5 ISA ch.8 (group0: count/lds/global/type=2;
// group1: data_size=2B, tensor_dim0/1, tile_dim0/1, dim0_stride).
// ---------------------------------------------------------------------------
#if defined(__HIP_DEVICE_COMPILE__)
__device__ __forceinline__ void tdm_load_rows_to_lds(const h16* gsrc,
                                                     unsigned lds_addr,
                                                     int rows, int cols, int ld) {
  unsigned long long ga = (unsigned long long)(size_t)gsrc;
  tdm_u32x4 g0;
  g0[0] = 1u;                                            // count=1 (valid D#)
  g0[1] = lds_addr;                                      // LDS byte address
  g0[2] = (unsigned)(ga & 0xFFFFFFFFu);                  // global_addr[31:0]
  g0[3] = (unsigned)((ga >> 32) & 0x1FFFFFFu) | (2u << 30);  // [56:32] | type=2
  unsigned td0 = (unsigned)cols;                         // tensor_dim0
  unsigned td1 = (unsigned)rows;                         // tensor_dim1
  tdm_i32x8 g1;
  g1[0] = (int)(1u << 16);                               // data_size=1 -> 2 bytes
  g1[1] = (int)((td0 & 0xFFFFu) << 16);                  // tensor_dim0 lo
  g1[2] = (int)((td0 >> 16) | ((td1 & 0xFFFFu) << 16));  // dim0 hi | dim1 lo
  g1[3] = (int)((td1 >> 16) | ((unsigned)cols << 16));   // dim1 hi | tile_dim0
  g1[4] = (int)((unsigned)rows & 0xFFFFu);               // tile_dim1 | tile_dim2=0
  g1[5] = (int)(unsigned)ld;                             // tensor_dim0_stride lo
  g1[6] = 0;                                             // stride0 hi | stride1 lo
  g1[7] = 0;                                             // stride1 hi
  tdm_i32x4 z4 = {0, 0, 0, 0};
#if defined(__clang_major__) && __clang_major__ >= 23
  tdm_i32x8 z8 = {0, 0, 0, 0, 0, 0, 0, 0};
  __builtin_amdgcn_tensor_load_to_lds(g0, g1, z4, z4, z8, 0);
#else
  __builtin_amdgcn_tensor_load_to_lds(g0, g1, z4, z4, 0);
#endif
}
#endif

// ---------------------------------------------------------------------------
// 1) top-k(256) of route_logits, descending, ties -> lowest index (jax)
// ---------------------------------------------------------------------------
__global__ void topk_kernel(const float* __restrict__ logits,
                            int* __restrict__ idx_out, int V, int Kt) {
  __shared__ unsigned int taken[512];
  __shared__ float sval[256];
  __shared__ int   sidx[256];
  int t = threadIdx.x;
  for (int i = t; i < 512; i += 256) taken[i] = 0u;
  __syncthreads();
  for (int iter = 0; iter < Kt; ++iter) {
    float bv = -INFINITY; int bi = V;
    for (int i = t; i < V; i += 256) {
      if (taken[i >> 5] & (1u << (i & 31))) continue;
      float vv = logits[i];
      if (vv > bv) { bv = vv; bi = i; }
    }
    sval[t] = bv; sidx[t] = bi;
    __syncthreads();
    for (int off = 128; off > 0; off >>= 1) {
      if (t < off) {
        float ov = sval[t + off]; int oi = sidx[t + off];
        if (ov > sval[t] || (ov == sval[t] && oi < sidx[t])) {
          sval[t] = ov; sidx[t] = oi;
        }
      }
      __syncthreads();
    }
    if (t == 0) {
      idx_out[iter] = sidx[0];
      taken[sidx[0] >> 5] |= (1u << (sidx[0] & 31));
    }
    __syncthreads();
  }
}

// ---------------------------------------------------------------------------
// 2) fp32 -> f16 conversion (weights)
// ---------------------------------------------------------------------------
__global__ void cvt_f16_kernel(const float* __restrict__ a,
                               h16* __restrict__ o, int n) {
  int i = blockIdx.x * blockDim.x + threadIdx.x;
  if (i < n) o[i] = (h16)a[i];
}

// ---------------------------------------------------------------------------
// 3) gather + rms_norm. One block (K threads) per token row.
// ---------------------------------------------------------------------------
__global__ void gather_norm_kernel(const float* __restrict__ x,
                                   const int* __restrict__ idx,
                                   float* __restrict__ gathered,
                                   h16* __restrict__ gnorm, int V, int K) {
  __shared__ float red[256];
  int row = blockIdx.x, t = threadIdx.x;
  float val = x[(size_t)row * V + idx[t]];
  red[t] = val * val;
  __syncthreads();
  for (int off = 128; off > 0; off >>= 1) {
    if (t < off) red[t] += red[t + off];
    __syncthreads();
  }
  float inv = rsqrtf(red[0] / (float)K + EPS_RMS);
  gathered[(size_t)row * K + t] = val;
  gnorm[(size_t)row * K + t] = (h16)(val * inv);
}

// ---------------------------------------------------------------------------
// 4) generic WMMA GEMM: D[M,N] = A[M,Kd] * W[N,Kd]^T, one 16x16 tile per wave.
//    mode 0: store f16   mode 1: store f32 * (*scale)   mode 2: +bias, GELU, f16
// ---------------------------------------------------------------------------
__global__ void gemm_wmma_kernel(const h16* __restrict__ A,
                                 const h16* __restrict__ W,
                                 h16* __restrict__ outh,
                                 float* __restrict__ outf,
                                 int M, int N, int Kd, int mode,
                                 const float* __restrict__ scale_p,
                                 const float* __restrict__ bias) {
  int lane = threadIdx.x;
  int row0 = blockIdx.x * 16;
  int col0 = blockIdx.y * 16;
  v8f acc = {};
  for (int k0 = 0; k0 < Kd; k0 += 32) {
    v16h a = load_a_tile(A, Kd, row0, k0, lane);
    v16h b = load_bt_tile(W, Kd, col0, k0, lane);
    acc = wmma_f16(a, b, acc);
  }
  int col = col0 + (lane & 15);
  int rb = (lane >> 4) * 8;
  float sc = (mode == 1) ? scale_p[0] : 1.0f;
#pragma unroll
  for (int r = 0; r < 8; ++r) {
    int row = row0 + rb + r;
    float v = acc[r];
    if (mode == 0) {
      outh[(size_t)row * N + col] = (h16)v;
    } else if (mode == 1) {
      outf[(size_t)row * N + col] = v * sc;
    } else {
      v += bias[col];
      v = 0.5f * v * (1.0f + erff(v * 0.70710678118654752440f));  // exact gelu
      outh[(size_t)row * N + col] = (h16)v;
    }
  }
}

// ---------------------------------------------------------------------------
// 5) decay "attention": one wave per 16-row tile. scores = q k^T weighted by
//    w(t,s) = decay^(s-t-1) for s>t (else 0); retrieved = scores @ v.
//    V tiles (32 x 256 f16, 16 KB) are DMA'd into LDS by the Tensor Data
//    Mover, double-buffered: the TDM for step i+1 overlaps the 16 score
//    WMMAs of step i; s_wait_tensorcnt(1) exploits in-order TDM completion.
// ---------------------------------------------------------------------------
__global__ void attn_kernel(const h16* __restrict__ Q, const h16* __restrict__ Km,
                            const h16* __restrict__ Vm, h16* __restrict__ retr,
                            const float* __restrict__ decay_logit,
                            int T, int K) {
  __shared__ h16 sS[16 * 32];
  __shared__ h16 vbuf[2][32 * 256];   // double-buffered V tile (2 x 16 KB)
  int lane = threadIdx.x;
  int tilesPerBatch = T / 16;
  int b = blockIdx.x / tilesPerBatch;
  int t0 = (blockIdx.x % tilesPerBatch) * 16;
  const h16* q = Q + (size_t)b * T * K;
  const h16* k = Km + (size_t)b * T * K;
  const h16* v = Vm + (size_t)b * T * K;

  float decay = 1.0f / (1.0f + __expf(-decay_logit[0]));
  float logd = __logf(decay);

  // q A-fragments for the whole K=256 reduction (8 chunks of 32)
  v16h qa[8];
#pragma unroll
  for (int c = 0; c < 8; ++c) qa[c] = load_a_tile(q, K, t0, c * 32, lane);

  v8f acc[16];
#pragma unroll
  for (int d = 0; d < 16; ++d) acc[d] = (v8f){};

  unsigned vb_addr0 = (unsigned)(size_t)(&vbuf[0][0]);
  unsigned vb_addr1 = (unsigned)(size_t)(&vbuf[1][0]);

  int col = lane & 15;
  int rb = (lane >> 4) * 8;
  int sstart = t0 & ~31;  // strictly-future weights zero out s <= t
  int nsteps = (T - sstart) / 32;

#if defined(__HIP_DEVICE_COMPILE__)
  tdm_load_rows_to_lds(v + (size_t)sstart * K, vb_addr0, 32, K, K);
#endif

  for (int i = 0; i < nsteps; ++i) {
    int s0 = sstart + i * 32;
    int cur = i & 1;
    bool have_next = (i + 1) < nsteps;
#if defined(__HIP_DEVICE_COMPILE__)
    if (have_next)
      tdm_load_rows_to_lds(v + (size_t)(s0 + 32) * K, cur ? vb_addr0 : vb_addr1,
                           32, K, K);
#endif
    if (have_next)  // hint next K tile rows into cache (global_prefetch_b8)
      __builtin_prefetch(k + (size_t)(s0 + 32 + lane) * K, 0, 1);

    v8f sc0 = {}, sc1 = {};
#pragma unroll
    for (int c = 0; c < 8; ++c) {
      v16h kb = load_bt_tile(k, K, s0, c * 32, lane);
      sc0 = wmma_f16(qa[c], kb, sc0);
    }
#pragma unroll
    for (int c = 0; c < 8; ++c) {
      v16h kb = load_bt_tile(k, K, s0 + 16, c * 32, lane);
      sc1 = wmma_f16(qa[c], kb, sc1);
    }
    // weight in C/D layout, stage into LDS row-major 16x32
#pragma unroll
    for (int r = 0; r < 8; ++r) {
      int t = t0 + rb + r;
      {
        int s = s0 + col;
        float w = (s > t) ? __expf(logd * (float)(s - t - 1)) : 0.0f;
        sS[(rb + r) * 32 + col] = (h16)(sc0[r] * w);
      }
      {
        int s = s0 + 16 + col;
        float w = (s > t) ? __expf(logd * (float)(s - t - 1)) : 0.0f;
        sS[(rb + r) * 32 + 16 + col] = (h16)(sc1[r] * w);
      }
    }
    __syncthreads();
    v16h sa = load_a_tile(sS, 32, 0, 0, lane);

    // wait for the current V tile (two in flight -> cnt<=1 means ours landed)
#if defined(__HIP_DEVICE_COMPILE__)
    if (have_next) __builtin_amdgcn_s_wait_tensorcnt(1);
    else           __builtin_amdgcn_s_wait_tensorcnt(0);
#endif

    int hh = lane >> 4;
#pragma unroll
    for (int d = 0; d < 16; ++d) {
      v16h vb;
#pragma unroll
      for (int e = 0; e < 16; ++e)
        vb[e] = vbuf[cur][(hh * 16 + e) * 256 + d * 16 + col];
      acc[d] = wmma_f16(sa, vb, acc[d]);
    }
    __syncthreads();
  }
  h16* out = retr + (size_t)b * T * K;
#pragma unroll
  for (int d = 0; d < 16; ++d)
#pragma unroll
    for (int r = 0; r < 8; ++r)
      out[(size_t)(t0 + rb + r) * K + d * 16 + col] = (h16)acc[d][r];
}

// ---------------------------------------------------------------------------
// 6) residual (+ optional rms_norm): nb = base + (*scale)*add
// ---------------------------------------------------------------------------
__global__ void residual_norm_kernel(const float* __restrict__ base,
                                     const float* __restrict__ add,
                                     const float* __restrict__ scale,
                                     float* __restrict__ nb,
                                     h16* __restrict__ gnorm,
                                     int K, int doNorm) {
  __shared__ float red[256];
  int row = blockIdx.x, t = threadIdx.x;
  size_t i = (size_t)row * K + t;
  float val = base[i] + scale[0] * add[i];
  nb[i] = val;
  if (doNorm) {
    red[t] = val * val;
    __syncthreads();
    for (int off = 128; off > 0; off >>= 1) {
      if (t < off) red[t] += red[t + off];
      __syncthreads();
    }
    float inv = rsqrtf(red[0] / (float)K + EPS_RMS);
    gnorm[i] = (h16)(val * inv);
  }
}

// ---------------------------------------------------------------------------
// 7) bulk copy (float4) and column scatter
// ---------------------------------------------------------------------------
__global__ void copy4_kernel(const float4* __restrict__ src,
                             float4* __restrict__ dst, size_t n4) {
  size_t i = blockIdx.x * (size_t)blockDim.x + threadIdx.x;
  size_t stride = (size_t)gridDim.x * blockDim.x;
  for (; i < n4; i += stride) dst[i] = src[i];
}

__global__ void scatter_kernel(const float* __restrict__ g3,
                               const int* __restrict__ idx,
                               float* __restrict__ out, int V, int K) {
  int row = blockIdx.x, t = threadIdx.x;
  out[(size_t)row * V + idx[t]] = g3[(size_t)row * K + t];
}

// ---------------------------------------------------------------------------
// host launcher
// ---------------------------------------------------------------------------
extern "C" void kernel_launch(void* const* d_in, const int* in_sizes, int n_in,
                              void* d_out, int out_size, void* d_ws, size_t ws_size,
                              hipStream_t stream) {
  const float* x            = (const float*)d_in[0];
  const float* route_logits = (const float*)d_in[1];
  const float* q_w          = (const float*)d_in[2];
  const float* k_w          = (const float*)d_in[3];
  const float* v_w          = (const float*)d_in[4];
  const float* o_w          = (const float*)d_in[5];
  const float* decay_logit  = (const float*)d_in[6];
  const float* mem_out_sc   = (const float*)d_in[7];
  const float* down_w       = (const float*)d_in[8];
  const float* up_w         = (const float*)d_in[9];
  const float* mlp_bias     = (const float*)d_in[10];
  const float* mlp_out_sc   = (const float*)d_in[11];
  const float* mem_scale    = (const float*)d_in[12];
  const float* mlp_scale    = (const float*)d_in[13];
  float* out = (float*)d_out;

  const int V = in_sizes[1];         // 16384
  const int K = 256;                 // q_w is K x K
  const int INNER = in_sizes[10];    // 512
  const int T = 2048;
  const int BT = in_sizes[0] / V;    // B*T = 4096

  char* p = (char*)d_ws;
  auto alloc = [&](size_t bytes) -> void* {
    void* r = (void*)p;
    p += (bytes + 255) & ~(size_t)255;
    return r;
  };
  int*  idx   = (int*)alloc(K * sizeof(int));
  h16*  wq    = (h16*)alloc((size_t)K * K * 2);
  h16*  wk    = (h16*)alloc((size_t)K * K * 2);
  h16*  wv    = (h16*)alloc((size_t)K * K * 2);
  h16*  wo    = (h16*)alloc((size_t)K * K * 2);
  h16*  wdn   = (h16*)alloc((size_t)INNER * K * 2);
  h16*  wup   = (h16*)alloc((size_t)K * INNER * 2);
  float* gath  = (float*)alloc((size_t)BT * K * 4);
  h16*  g1    = (h16*)alloc((size_t)BT * K * 2);
  h16*  qh    = (h16*)alloc((size_t)BT * K * 2);
  h16*  kh    = (h16*)alloc((size_t)BT * K * 2);
  h16*  vh    = (h16*)alloc((size_t)BT * K * 2);
  h16*  retr  = (h16*)alloc((size_t)BT * K * 2);
  float* memo  = (float*)alloc((size_t)BT * K * 4);
  float* gath2 = (float*)alloc((size_t)BT * K * 4);
  h16*  g2    = (h16*)alloc((size_t)BT * K * 2);
  h16*  hmid  = (h16*)alloc((size_t)BT * INNER * 2);
  float* mlpo  = (float*)alloc((size_t)BT * K * 4);
  float* gath3 = (float*)alloc((size_t)BT * K * 4);
  (void)ws_size; (void)n_in; (void)out_size;

  // 1) routing indices
  topk_kernel<<<1, 256, 0, stream>>>(route_logits, idx, V, K);

  // 2) weight conversion to f16
  auto cvt = [&](const float* src, h16* dst, int n) {
    cvt_f16_kernel<<<(n + 255) / 256, 256, 0, stream>>>(src, dst, n);
  };
  cvt(q_w, wq, K * K); cvt(k_w, wk, K * K);
  cvt(v_w, wv, K * K); cvt(o_w, wo, K * K);
  cvt(down_w, wdn, INNER * K); cvt(up_w, wup, K * INNER);

  // 3) gather + rms_norm
  gather_norm_kernel<<<BT, K, 0, stream>>>(x, idx, gath, g1, V, K);

  // 4) q/k/v projections (f16 out)
  dim3 gKK(BT / 16, K / 16);
  gemm_wmma_kernel<<<gKK, 32, 0, stream>>>(g1, wq, qh, nullptr, BT, K, K, 0, nullptr, nullptr);
  gemm_wmma_kernel<<<gKK, 32, 0, stream>>>(g1, wk, kh, nullptr, BT, K, K, 0, nullptr, nullptr);
  gemm_wmma_kernel<<<gKK, 32, 0, stream>>>(g1, wv, vh, nullptr, BT, K, K, 0, nullptr, nullptr);

  // 5) decay attention (WMMA + TDM double-buffered V tiles)
  attn_kernel<<<BT / 16, 32, 0, stream>>>(qh, kh, vh, retr, decay_logit, T, K);

  // 6) o-projection * mem_out_scale (f32 out)
  gemm_wmma_kernel<<<gKK, 32, 0, stream>>>(retr, wo, nullptr, memo, BT, K, K, 1, mem_out_sc, nullptr);

  // 7) residual + rms_norm
  residual_norm_kernel<<<BT, K, 0, stream>>>(gath, memo, mem_scale, gath2, g2, K, 1);

  // 8) MLP down (bias + exact gelu, f16 out)
  dim3 gKI(BT / 16, INNER / 16);
  gemm_wmma_kernel<<<gKI, 32, 0, stream>>>(g2, wdn, hmid, nullptr, BT, INNER, K, 2, nullptr, mlp_bias);

  // 9) MLP up * mlp_out_scale (f32 out)
  gemm_wmma_kernel<<<gKK, 32, 0, stream>>>(hmid, wup, nullptr, mlpo, BT, K, INNER, 1, mlp_out_sc, nullptr);

  // 10) final residual (no norm needed)
  residual_norm_kernel<<<BT, K, 0, stream>>>(gath2, mlpo, mlp_scale, gath3, nullptr, K, 0);

  // 11) out = x, then scatter gathered columns back
  size_t n4 = ((size_t)BT * V) / 4;
  copy4_kernel<<<8192, 256, 0, stream>>>((const float4*)x, (float4*)out, n4);
  scatter_kernel<<<BT, K, 0, stream>>>(gath3, idx, out, V, K);
}
